// GraphEncoder_64957085385269
// MI455X (gfx1250) — compile-verified, matrix operand order checked by reading, not run on hardware
//
#include <hip/hip_runtime.h>
#include <hip/hip_bf16.h>

#define NN 50000
#define EE 800000
#define ETOT (EE + NN)
#define HHD 4
#define CC 64
#define HC 256
#define FEATD 128
#define EDIM 16
#define GG 256
#define NEG_SLOPE 0.2f
#define BN_EPS 1e-5f

typedef __attribute__((ext_vector_type(16))) _Float16 v16h;
typedef __attribute__((ext_vector_type(8)))  _Float16 v8h;
typedef __attribute__((ext_vector_type(4)))  _Float16 v4h;
typedef __attribute__((ext_vector_type(8)))  float    v8f;

// ---------- helpers ----------
__device__ __forceinline__ unsigned encf(float x) {
  unsigned u = __float_as_uint(x);
  return (u & 0x80000000u) ? ~u : (u | 0x80000000u);
}
__device__ __forceinline__ float decf(unsigned u) {
  unsigned v = (u & 0x80000000u) ? (u & 0x7FFFFFFFu) : ~u;
  return __uint_as_float(v);
}

__global__ void k_zero(unsigned* __restrict__ p, size_t n) {
  size_t t = (size_t)blockIdx.x * blockDim.x + threadIdx.x;
  if (t < n) p[t] = 0u;
}

// ---------- edge-attr mean (partial sums per dim) ----------
__global__ void k_easum(const float* __restrict__ ea, float* __restrict__ acc) {
  __shared__ float lds[256];
  int d = threadIdx.x & 15, grp = threadIdx.x >> 4;
  float s = 0.f;
  for (int e = blockIdx.x * 16 + grp; e < EE; e += gridDim.x * 16)
    s += ea[(size_t)e * EDIM + d];
  lds[threadIdx.x] = s;
  __syncthreads();
  if (threadIdx.x < 16) {
    float t = 0.f;
    #pragma unroll
    for (int g = 0; g < 16; g++) t += lds[g * 16 + threadIdx.x];
    atomicAdd(&acc[threadIdx.x], t);
  }
}

// w_e[d][h] = sum_c We0[d, h*64+c]*att_e0[h,c];  mean16 = acc/E
__global__ void k_we(const float* __restrict__ We, const float* __restrict__ ae,
                     const float* __restrict__ acc, float* __restrict__ w_e,
                     float* __restrict__ mean16) {
  int t = threadIdx.x;           // 64 threads
  int d = t >> 2, hh = t & 3;
  float s = 0.f;
  for (int c = 0; c < CC; c++) s += We[d * HC + hh * CC + c] * ae[hh * CC + c];
  w_e[t] = s;
  if (t < EDIM) mean16[t] = acc[t] * (1.0f / EE);
}

__global__ void k_alpha_e(const float* __restrict__ ea, const float* __restrict__ w_e,
                          const float* __restrict__ mean16, float* __restrict__ aE) {
  int e = blockIdx.x * blockDim.x + threadIdx.x;
  if (e >= ETOT) return;
  float v[EDIM];
  if (e < EE) {
    #pragma unroll
    for (int d = 0; d < EDIM; d++) v[d] = ea[(size_t)e * EDIM + d];
  } else {
    #pragma unroll
    for (int d = 0; d < EDIM; d++) v[d] = mean16[d];
  }
  #pragma unroll
  for (int hh = 0; hh < HHD; hh++) {
    float s = 0.f;
    #pragma unroll
    for (int d = 0; d < EDIM; d++) s += v[d] * w_e[d * 4 + hh];
    aE[(size_t)e * HHD + hh] = s;
  }
}

// ---------- precision prep for WMMA GEMM ----------
// Wt[col][k] = (f16) W[k][col]  (transposed, f16)
__global__ void k_wconv(const float* __restrict__ W, _Float16* __restrict__ Wt, int K) {
  int k = blockIdx.x, col = threadIdx.x;     // grid = K, block = 256
  Wt[(size_t)col * K + k] = (_Float16)W[(size_t)k * HC + col];
}

// Xh = (f16) X, 4 elements per thread
__global__ void k_xconv(const float* __restrict__ X, _Float16* __restrict__ Xh, size_t n4) {
  size_t t = (size_t)blockIdx.x * blockDim.x + threadIdx.x;
  if (t >= n4) return;
  float4 v = ((const float4*)X)[t];
  v4h o;
  o[0] = (_Float16)v.x; o[1] = (_Float16)v.y; o[2] = (_Float16)v.z; o[3] = (_Float16)v.w;
  ((v4h*)Xh)[t] = o;
}

// ---------- WMMA GEMM: Hout[N,256] = Xh[N,K] * W[K,256] ----------
// 8 waves/block: 4 row-tiles x 2 column-halves; 8 persistent accumulators per
// wave (64 VGPRs, no spill). All waves run the full loop with EXEC all-ones
// (WMMA requirement); only the final stores are bounds-guarded. Tail-wave
// overreads land inside the Wt carve (valid workspace).
__global__ __launch_bounds__(256, 1)
void k_gemm(const _Float16* __restrict__ Xh, const _Float16* __restrict__ Wt,
            float* __restrict__ Hout, int K) {
  __shared__ _Float16 lWt[32 * HC];          // 16KB slab: [col][32]
  int tid = threadIdx.x;
  int wave = tid >> 5, lane = tid & 31;
  int half = lane >> 4, sub = lane & 15;
  int tmw = wave >> 1, tnh = wave & 1;       // row tile in block, column half
  int tm = blockIdx.x * 4 + tmw;
  size_t rowK = (size_t)(tm * 16 + sub) * (size_t)K;
  v8f acc[8];
  #pragma unroll
  for (int t = 0; t < 8; t++) acc[t] = (v8f){};

  for (int k0 = 0; k0 < K; k0 += 32) {
    __syncthreads();
    // cooperative stage: 1024 x 16B chunks (256 cols x 32 k-halfs)
    #pragma unroll
    for (int it = 0; it < 4; it++) {
      int c = tid + it * 256;
      int col = c >> 2, part = c & 3;
      *(v8h*)(lWt + col * 32 + part * 8) =
          *(const v8h*)(Wt + (size_t)col * K + k0 + part * 8);
    }
    __syncthreads();

    int kb = k0 + half * 8;
    v16h a;
    {
      v8h a0 = *(const v8h*)(Xh + rowK + kb);
      v8h a1 = *(const v8h*)(Xh + rowK + kb + 16);
      #pragma unroll
      for (int i = 0; i < 8; i++) { a[i] = a0[i]; a[8 + i] = a1[i]; }
    }
    #pragma unroll
    for (int t = 0; t < 8; t++) {
      int col = (tnh * 8 + t) * 16 + sub;
      const _Float16* bp = lWt + col * 32 + half * 8;
      v8h b0 = *(const v8h*)bp;
      v8h b1 = *(const v8h*)(bp + 16);
      v16h b;
      #pragma unroll
      for (int i = 0; i < 8; i++) { b[i] = b0[i]; b[8 + i] = b1[i]; }
      acc[t] = __builtin_amdgcn_wmma_f32_16x16x32_f16(false, a, false, b,
                                                      (short)0, acc[t], false, false);
    }
  }
  if (tm >= NN / 16) return;                 // after all WMMAs; stores only
  // D layout: VGPR v -> M = v + 8*half, N = sub
  #pragma unroll
  for (int t = 0; t < 8; t++) {
    float* op = Hout + (size_t)(tm * 16 + half * 8) * HC + (tnh * 8 + t) * 16 + sub;
    #pragma unroll
    for (int v = 0; v < 8; v++) op[(size_t)v * HC] = acc[t][v];
  }
}

// ---------- attention scores per (node, head) ----------
__global__ void k_scores(const float* __restrict__ Hb, const float* __restrict__ as,
                         const float* __restrict__ ad, float* __restrict__ ssrc,
                         float* __restrict__ sdst) {
  int t = blockIdx.x * blockDim.x + threadIdx.x;
  if (t >= NN * HHD) return;
  int n = t >> 2, hh = t & 3;
  const float* hr = Hb + (size_t)n * HC + hh * CC;
  float s0 = 0.f, s1 = 0.f;
  #pragma unroll 8
  for (int c = 0; c < CC; c++) {
    float v = hr[c];
    s0 += v * as[hh * CC + c];
    s1 += v * ad[hh * CC + c];
  }
  ssrc[t] = s0;
  sdst[t] = s1;
}

// ---------- edge softmax pass 1: leaky-relu + segment max (ordered-uint atomic) ----------
__global__ void k_attn_max(const int* __restrict__ ei, const float* __restrict__ ssrc,
                           const float* __restrict__ sdst, const float* __restrict__ aE,
                           float* __restrict__ ealpha, unsigned* __restrict__ amax) {
  int t = blockIdx.x * blockDim.x + threadIdx.x;
  if (t >= ETOT * HHD) return;
  int e = t >> 2, hh = t & 3;
  int src = (e < EE) ? ei[e] : (e - EE);
  int dst = (e < EE) ? ei[EE + e] : (e - EE);
  float a = ssrc[src * 4 + hh] + sdst[dst * 4 + hh];
  if (aE) a += aE[t];
  a = (a > 0.f) ? a : NEG_SLOPE * a;
  ealpha[t] = a;
  atomicMax(&amax[dst * 4 + hh], encf(a));
}

// ---------- edge softmax pass 2: exp + segment sum ----------
__global__ void k_attn_exp(const int* __restrict__ ei, float* __restrict__ ealpha,
                           const unsigned* __restrict__ amax, float* __restrict__ denom) {
  int t = blockIdx.x * blockDim.x + threadIdx.x;
  if (t >= ETOT * HHD) return;
  int e = t >> 2, hh = t & 3;
  int dst = (e < EE) ? ei[EE + e] : (e - EE);
  float m = decf(amax[dst * 4 + hh]);
  float v = __expf(ealpha[t] - m);
  ealpha[t] = v;
  atomicAdd(&denom[dst * 4 + hh], v);
}

// ---------- aggregation: one wave per edge, lane-strided 256 columns ----------
__global__ void k_aggregate(const int* __restrict__ ei, const float* __restrict__ Hb,
                            const float* __restrict__ ealpha, const float* __restrict__ denom,
                            float* __restrict__ outb) {
  int e = blockIdx.x * 8 + (threadIdx.x >> 5);
  if (e >= ETOT) return;
  int lane = threadIdx.x & 31;
  int src = (e < EE) ? ei[e] : (e - EE);
  int dst = (e < EE) ? ei[EE + e] : (e - EE);
  const float* hs = Hb + (size_t)src * HC;
  __builtin_prefetch(hs + lane, 0, 0);       // global_prefetch of gathered row
  float coef[4];
  #pragma unroll
  for (int hh = 0; hh < 4; hh++)
    coef[hh] = ealpha[(size_t)e * 4 + hh] / (denom[dst * 4 + hh] + 1e-16f);
  float* od = outb + (size_t)dst * HC;
  #pragma unroll
  for (int j = 0; j < 8; j++) {
    int colj = lane + 32 * j;                // head = j>>1 (uniform per j)
    atomicAdd(od + colj, coef[j >> 1] * hs[colj]);
  }
}

// ---------- batch-norm stats ----------
__global__ void k_bnstats(const float* __restrict__ ob, float* __restrict__ sumc,
                          float* __restrict__ sqc) {
  int col = threadIdx.x;                     // 256
  float s = 0.f, q = 0.f;
  for (int n = blockIdx.x; n < NN; n += gridDim.x) {
    float v = ob[(size_t)n * HC + col];
    s += v;
    q += v * v;
  }
  atomicAdd(&sumc[col], s);
  atomicAdd(&sqc[col], q);
}

__global__ void k_bnfin(const float* __restrict__ sumc, const float* __restrict__ sqc,
                        const float* __restrict__ g, const float* __restrict__ b,
                        float* __restrict__ sc, float* __restrict__ sh) {
  int c = threadIdx.x;
  float mu = sumc[c] * (1.0f / NN);
  float var = sqc[c] * (1.0f / NN) - mu * mu;
  float s = g[c] * rsqrtf(var + BN_EPS);
  sc[c] = s;
  sh[c] = b[c] - mu * s;                     // bias cancels under BN, omitted
}

__global__ void k_bnrelu(const float* __restrict__ ob, const float* __restrict__ sc,
                         const float* __restrict__ sh, float* __restrict__ xn) {
  size_t t = (size_t)blockIdx.x * blockDim.x + threadIdx.x;
  if (t >= (size_t)NN * HC) return;
  int c = (int)(t & 255);
  float y = fmaf(ob[t], sc[c], sh[c]);
  xn[t] = fmaxf(y, 0.f);
}

// ---------- global mean pool ----------
__global__ void k_count(const int* __restrict__ batch, float* __restrict__ counts) {
  int n = blockIdx.x * blockDim.x + threadIdx.x;
  if (n < NN) atomicAdd(&counts[batch[n]], 1.0f);
}

__global__ void k_sums(const float* __restrict__ x, const int* __restrict__ batch,
                       float* __restrict__ sums) {
  int n = blockIdx.x;                        // one block per node
  int c = threadIdx.x;                       // 256
  atomicAdd(&sums[(size_t)batch[n] * HC + c], x[(size_t)n * HC + c]);
}

__global__ void k_final(const float* __restrict__ sums, const float* __restrict__ counts,
                        float* __restrict__ out) {
  int t = blockIdx.x * blockDim.x + threadIdx.x;
  if (t >= GG * HC) return;
  int g = t >> 8;
  out[t] = sums[t] / fmaxf(counts[g], 1.0f);
}

extern "C" void kernel_launch(void* const* d_in, const int* in_sizes, int n_in,
                              void* d_out, int out_size, void* d_ws, size_t ws_size,
                              hipStream_t stream) {
  const float* x0    = (const float*)d_in[0];
  const int*   ei    = (const int*)d_in[1];
  const float* eatt  = (const float*)d_in[2];
  const int*   batch = (const int*)d_in[3];
  const float* W[3]     = {(const float*)d_in[4],  (const float*)d_in[10], (const float*)d_in[16]};
  const float* asrc[3]  = {(const float*)d_in[5],  (const float*)d_in[11], (const float*)d_in[17]};
  const float* adst[3]  = {(const float*)d_in[6],  (const float*)d_in[12], (const float*)d_in[18]};
  const float* gamma[3] = {(const float*)d_in[8],  (const float*)d_in[14], (const float*)d_in[20]};
  const float* beta[3]  = {(const float*)d_in[9],  (const float*)d_in[15], (const float*)d_in[21]};
  const float* We0  = (const float*)d_in[22];
  const float* atte = (const float*)d_in[23];
  float* out = (float*)d_out;

  // ---- workspace carve (~210 MB) ----
  float* p = (float*)d_ws;
  float* x_cur  = p; p += (size_t)NN * HC;
  float* Hb     = p; p += (size_t)NN * HC;
  float* outb   = p; p += (size_t)NN * HC;
  float* ealpha = p; p += (size_t)ETOT * HHD;
  float* alphaE = p; p += (size_t)ETOT * HHD;
  float* ssrc   = p; p += (size_t)NN * HHD;
  float* sdst   = p; p += (size_t)NN * HHD;
  unsigned* amax = (unsigned*)p; p += (size_t)NN * HHD;   // contiguous with denom
  float* denom  = p; p += (size_t)NN * HHD;
  float* sumc   = p; p += 256;                             // contiguous with sqc
  float* sqc    = p; p += 256;
  float* scb    = p; p += 256;
  float* shiftb = p; p += 256;
  float* acc16  = p; p += 16;
  float* mean16 = p; p += 16;
  float* w_e    = p; p += 64;
  float* counts = p; p += GG;                              // contiguous with sums
  float* sums   = p; p += (size_t)GG * HC;
  _Float16* Xh  = (_Float16*)p; p += (size_t)NN * HC / 2;  // f16 activations
  _Float16* Wt  = (_Float16*)p; p += (size_t)HC * HC / 2;  // f16 transposed weights

  auto blocks = [](size_t n, int bs) { return (unsigned)((n + bs - 1) / bs); };

  // edge-attention precompute
  k_zero<<<1, 256, 0, stream>>>((unsigned*)acc16, 16);
  k_easum<<<128, 256, 0, stream>>>(eatt, acc16);
  k_we<<<1, 64, 0, stream>>>(We0, atte, acc16, w_e, mean16);
  k_alpha_e<<<blocks(ETOT, 256), 256, 0, stream>>>(eatt, w_e, mean16, alphaE);

  for (int l = 0; l < 3; l++) {
    const float* xin = (l == 0) ? x0 : x_cur;
    int K = (l == 0) ? FEATD : HC;

    k_zero<<<blocks((size_t)NN * HHD * 2, 256), 256, 0, stream>>>(amax, (size_t)NN * HHD * 2);
    k_zero<<<blocks((size_t)NN * HC, 256), 256, 0, stream>>>((unsigned*)outb, (size_t)NN * HC);
    k_zero<<<1, 512, 0, stream>>>((unsigned*)sumc, 512);

    k_wconv<<<K, 256, 0, stream>>>(W[l], Wt, K);
    k_xconv<<<blocks((size_t)NN * K / 4, 256), 256, 0, stream>>>(xin, Xh, (size_t)NN * K / 4);
    k_gemm<<<blocks(NN / 16, 4), 256, 0, stream>>>(Xh, Wt, Hb, K);   // 782 blocks, 8 waves

    k_scores<<<blocks((size_t)NN * HHD, 256), 256, 0, stream>>>(Hb, asrc[l], adst[l], ssrc, sdst);
    k_attn_max<<<blocks((size_t)ETOT * HHD, 256), 256, 0, stream>>>(
        ei, ssrc, sdst, (l == 0) ? alphaE : nullptr, ealpha, amax);
    k_attn_exp<<<blocks((size_t)ETOT * HHD, 256), 256, 0, stream>>>(ei, ealpha, amax, denom);
    k_aggregate<<<blocks(ETOT, 8), 256, 0, stream>>>(ei, Hb, ealpha, denom, outb);
    k_bnstats<<<512, 256, 0, stream>>>(outb, sumc, sqc);
    k_bnfin<<<1, 256, 0, stream>>>(sumc, sqc, gamma[l], beta[l], scb, shiftb);
    k_bnrelu<<<blocks((size_t)NN * HC, 256), 256, 0, stream>>>(outb, scb, shiftb, x_cur);
  }

  // global mean pool
  k_zero<<<blocks((size_t)GG * (HC + 1), 256), 256, 0, stream>>>((unsigned*)counts,
                                                                 (size_t)GG * (HC + 1));
  k_count<<<blocks(NN, 256), 256, 0, stream>>>(batch, counts);
  k_sums<<<NN, 256, 0, stream>>>(x_cur, batch, sums);
  k_final<<<blocks((size_t)GG * HC, 256), 256, 0, stream>>>(sums, counts, out);
}
